// ResidualAttentionBlock_59167469470411
// MI455X (gfx1250) — compile-verified, matrix operand order checked by reading, not run on hardware
//
#include <hip/hip_runtime.h>

typedef __attribute__((ext_vector_type(16))) _Float16 v16h;
typedef __attribute__((ext_vector_type(8)))  _Float16 v8h;
typedef __attribute__((ext_vector_type(8)))  float    v8f;

#define WMMA_F16(a, b, c) \
  __builtin_amdgcn_wmma_f32_16x16x32_f16(false, (a), false, (b), (short)0, (c), false, false)

// ---------------------------------------------------------------------------
// Fragment load from GLOBAL: 16-bit A or B fragment (16 rows x 32 K) from
// (rows, K) row-major storage. Per ISA layout: lane L holds row (L&15); its
// 16 halves are K[kb..kb+7] and K[kb+16..kb+23] with kb = k0 + (L>>4)*8.
// ---------------------------------------------------------------------------
__device__ inline v16h load_frag(const _Float16* __restrict__ base, int ld,
                                 int row0, int k0) {
  int lane = threadIdx.x & 31;
  const _Float16* p = base + (size_t)(row0 + (lane & 15)) * ld + k0 + ((lane >> 4) << 3);
  v8h lo = *(const v8h*)p;
  v8h hi = *(const v8h*)(p + 16);
  return __builtin_shufflevector(lo, hi, 0,1,2,3,4,5,6,7,8,9,10,11,12,13,14,15);
}

// ---------------------------------------------------------------------------
// Fragment load from LDS tile with padded row stride of 40 halves (80 B):
// 16 fragment rows hit 16 distinct bank groups -> conflict-free ds_load_b128.
// ---------------------------------------------------------------------------
#define LDS_STRIDE 40

__device__ inline v16h lds_frag(const _Float16* lds, int row0) {
  int lane = threadIdx.x & 31;
  const _Float16* p = lds + (row0 + (lane & 15)) * LDS_STRIDE + ((lane >> 4) << 3);
  v8h lo = *(const v8h*)p;
  v8h hi = *(const v8h*)(p + 16);
  return __builtin_shufflevector(lo, hi, 0,1,2,3,4,5,6,7,8,9,10,11,12,13,14,15);
}

// Stage a 128-row x 32-col f16 tile from global (row-major, leading dim ld)
// into LDS (row stride LDS_STRIDE). 256 threads, coalesced 16B chunks:
// threads t..t+3 cover one contiguous 64B row segment.
__device__ inline void stage_tile(const _Float16* __restrict__ g, int ld,
                                  int row0, int k0, _Float16* __restrict__ lds) {
  int t = threadIdx.x;
#pragma unroll
  for (int i = 0; i < 2; i++) {
    int c = t + i * 256;          // chunk 0..511 (8 halves each)
    int row = c >> 2;
    int piece = (c & 3) << 3;
    v8h v = *(const v8h*)(g + (size_t)(row0 + row) * ld + k0 + piece);
    *(v8h*)(lds + row * LDS_STRIDE + piece) = v;
  }
}

// ---------------------------------------------------------------------------
// Elementwise weight converters
// ---------------------------------------------------------------------------
__global__ void cvt_f32_to_f16(const float* __restrict__ in,
                               _Float16* __restrict__ out, int n) {
  int i = blockIdx.x * blockDim.x + threadIdx.x;
  if (i < n) out[i] = (_Float16)in[i];
}

// in: (R, C) f32 row-major  ->  out: (C, R) f16 row-major
__global__ void transpose_to_f16(const float* __restrict__ in,
                                 _Float16* __restrict__ out, int R, int C) {
  int i = blockIdx.x * blockDim.x + threadIdx.x;
  if (i < R * C) {
    int r = i / C, c = i % C;
    out[(size_t)c * R + r] = (_Float16)in[i];
  }
}

// ---------------------------------------------------------------------------
// LayerNorm over rows of 1024, f32 in -> f16 out (feeds WMMA GEMMs)
// ---------------------------------------------------------------------------
__global__ __launch_bounds__(256) void layernorm_to_f16(
    const float* __restrict__ x, const float* __restrict__ gamma,
    const float* __restrict__ beta, _Float16* __restrict__ out) {
  const int row = blockIdx.x;
  const float* xr = x + (size_t)row * 1024;
  float v[4], s = 0.f, sq = 0.f;
#pragma unroll
  for (int i = 0; i < 4; i++) {
    v[i] = xr[threadIdx.x + i * 256];
    s += v[i];
    sq += v[i] * v[i];
  }
#pragma unroll
  for (int m = 16; m >= 1; m >>= 1) {
    s  += __shfl_xor(s, m, 32);
    sq += __shfl_xor(sq, m, 32);
  }
  __shared__ float ss[8], ssq[8];
  int w = threadIdx.x >> 5;
  if ((threadIdx.x & 31) == 0) { ss[w] = s; ssq[w] = sq; }
  __syncthreads();
  if (threadIdx.x < 32) {
    float a  = (threadIdx.x < 8) ? ss[threadIdx.x]  : 0.f;
    float aq = (threadIdx.x < 8) ? ssq[threadIdx.x] : 0.f;
#pragma unroll
    for (int m = 4; m >= 1; m >>= 1) {
      a  += __shfl_xor(a, m, 32);
      aq += __shfl_xor(aq, m, 32);
    }
    if (threadIdx.x == 0) { ss[0] = a; ssq[0] = aq; }
  }
  __syncthreads();
  float mean = ss[0] * (1.f / 1024.f);
  float var  = ssq[0] * (1.f / 1024.f) - mean * mean;
  float inv  = rsqrtf(var + 1e-5f);
#pragma unroll
  for (int i = 0; i < 4; i++) {
    int c = threadIdx.x + i * 256;
    out[(size_t)row * 1024 + c] = (_Float16)((v[i] - mean) * inv * gamma[c] + beta[c]);
  }
}

// ---------------------------------------------------------------------------
// LDS-staged WMMA GEMM: C[M,N] = A16[M,K] * Bt16[N,K]^T (+bias)(+relu)(+res)
// Workgroup tile 128x128, K-step 32, double-buffered LDS (40 KB).
// 8 waves, each owns a 64x32 sub-tile = 4x2 WMMA C tiles (8 WMMA / K-step).
// Each tile byte is fetched from L2 exactly once: 16 KB / 1.05 MFLOP per
// workgroup K-step = 64 FLOP/byte of L2 traffic.
// ---------------------------------------------------------------------------
template <bool BIAS, bool RELU, bool RES, bool OUT32, bool OUT16>
__global__ __launch_bounds__(256) void gemm_wmma(
    const _Float16* __restrict__ A, const _Float16* __restrict__ Bt,
    const float* __restrict__ bias, const float* __restrict__ resid,
    float* __restrict__ C32, _Float16* __restrict__ C16,
    int M, int N, int K) {
  const int w    = threadIdx.x >> 5;
  const int lane = threadIdx.x & 31;
  const int wm = w >> 2;          // 0..1 -> 64-row slice
  const int wn = w & 3;           // 0..3 -> 32-col slice
  const int mblk = blockIdx.y * 128;
  const int nblk = blockIdx.x * 128;

  __shared__ __align__(16) _Float16 As[2][128 * LDS_STRIDE];
  __shared__ __align__(16) _Float16 Bs[2][128 * LDS_STRIDE];

  v8f acc[4][2];
#pragma unroll
  for (int mt = 0; mt < 4; mt++)
#pragma unroll
    for (int nt = 0; nt < 2; nt++) acc[mt][nt] = {};

  stage_tile(A,  K, mblk, 0, As[0]);
  stage_tile(Bt, K, nblk, 0, Bs[0]);
  __syncthreads();

  int buf = 0;
  for (int k0 = 0; k0 < K; k0 += 32) {
    if (k0 + 32 < K) {
      stage_tile(A,  K, mblk, k0 + 32, As[buf ^ 1]);
      stage_tile(Bt, K, nblk, k0 + 32, Bs[buf ^ 1]);
    }
    v16h a[4], b[2];
#pragma unroll
    for (int mt = 0; mt < 4; mt++) a[mt] = lds_frag(As[buf], wm * 64 + mt * 16);
#pragma unroll
    for (int nt = 0; nt < 2; nt++) b[nt] = lds_frag(Bs[buf], wn * 32 + nt * 16);
#pragma unroll
    for (int mt = 0; mt < 4; mt++)
#pragma unroll
      for (int nt = 0; nt < 2; nt++)
        acc[mt][nt] = WMMA_F16(a[mt], b[nt], acc[mt][nt]);
    __syncthreads();
    buf ^= 1;
  }

  const int col   = lane & 15;
  const int rbase = (lane >> 4) << 3;
#pragma unroll
  for (int mt = 0; mt < 4; mt++) {
#pragma unroll
    for (int nt = 0; nt < 2; nt++) {
      int n = nblk + wn * 32 + nt * 16 + col;
      float bv = BIAS ? bias[n] : 0.f;
#pragma unroll
      for (int r = 0; r < 8; r++) {
        int m = mblk + wm * 64 + mt * 16 + rbase + r;
        float vv = acc[mt][nt][r] + bv;
        if (RELU) vv = fmaxf(vv, 0.f);
        if (RES) vv += resid[(size_t)m * N + n];
        if (OUT32) C32[(size_t)m * N + n] = vv;
        if (OUT16) C16[(size_t)m * N + n] = (_Float16)vv;
      }
    }
  }
}

// ---------------------------------------------------------------------------
// Scatter qkv (BS, 3*d) f16 -> q/k as (B*H, S, dh), v transposed (B*H, dh, S)
// ---------------------------------------------------------------------------
__global__ void reorder_qkv(const _Float16* __restrict__ qkv,
                            _Float16* __restrict__ q, _Float16* __restrict__ k,
                            _Float16* __restrict__ vT) {
  int i = blockIdx.x * blockDim.x + threadIdx.x;
  if (i >= 4096 * 3072) return;
  int row = i / 3072, n = i % 3072;
  int t = n >> 10;
  int hn = n & 1023;
  int h = hn >> 6, e = hn & 63;
  int b = row >> 10, s = row & 1023;
  _Float16 val = qkv[i];
  int bh = b * 16 + h;
  if (t == 0)      q[((size_t)bh * 1024 + s) * 64 + e] = val;
  else if (t == 1) k[((size_t)bh * 1024 + s) * 64 + e] = val;
  else             vT[((size_t)bh * 64 + e) * 1024 + s] = val;
}

// ---------------------------------------------------------------------------
// Flash attention: one wave per 16 query rows of one (b,h).
// Online softmax over WMMA C-layout; P staged through LDS to A-layout.
// Output written (B, S, H*dh) f16 for the out-projection GEMM.
// ---------------------------------------------------------------------------
__global__ __launch_bounds__(128) void attention_kernel(
    const _Float16* __restrict__ q, const _Float16* __restrict__ k,
    const _Float16* __restrict__ vT, _Float16* __restrict__ o) {
  constexpr int S = 1024, DH = 64;
  const int wave = threadIdx.x >> 5;
  const int lane = threadIdx.x & 31;
  const int bh = blockIdx.y;
  const int qrow0 = blockIdx.x * 64 + wave * 16;

  const _Float16* qb = q  + (size_t)bh * S * DH;
  const _Float16* kb = k  + (size_t)bh * S * DH;
  const _Float16* vb = vT + (size_t)bh * DH * S;

  __shared__ __align__(16) _Float16 pbuf[4][16 * 32];
  _Float16* pl = pbuf[wave];

  const v16h qa0 = load_frag(qb, DH, qrow0, 0);
  const v16h qa1 = load_frag(qb, DH, qrow0, 32);

  float mrow[8], lrow[8];
  v8f oacc[4];
#pragma unroll
  for (int r = 0; r < 8; r++) { mrow[r] = -1e30f; lrow[r] = 0.f; }
#pragma unroll
  for (int t = 0; t < 4; t++) oacc[t] = {};

  const int col   = lane & 15;
  const int rbase = (lane >> 4) << 3;

  for (int jb = 0; jb < S; jb += 32) {
    // scores for 32 keys: two 16x16 tiles, K = dh = 64 = 2 WMMA steps
    v8f s0 = {}, s1 = {};
    {
      v16h k0a = load_frag(kb, DH, jb, 0);
      v16h k0b = load_frag(kb, DH, jb, 32);
      s0 = WMMA_F16(qa0, k0a, s0);
      s0 = WMMA_F16(qa1, k0b, s0);
      v16h k1a = load_frag(kb, DH, jb + 16, 0);
      v16h k1b = load_frag(kb, DH, jb + 16, 32);
      s1 = WMMA_F16(qa0, k1a, s1);
      s1 = WMMA_F16(qa1, k1b, s1);
    }
    // online softmax update per row (row r lives across a 16-lane half)
#pragma unroll
    for (int r = 0; r < 8; r++) {
      float a0 = s0[r] * 0.125f;  // 1/sqrt(64)
      float a1 = s1[r] * 0.125f;
      float mx = fmaxf(a0, a1);
#pragma unroll
      for (int m = 8; m >= 1; m >>= 1) mx = fmaxf(mx, __shfl_xor(mx, m, 32));
      float mnew  = fmaxf(mrow[r], mx);
      float alpha = __expf(mrow[r] - mnew);
      float p0 = __expf(a0 - mnew);
      float p1 = __expf(a1 - mnew);
      float rs = p0 + p1;
#pragma unroll
      for (int m = 8; m >= 1; m >>= 1) rs += __shfl_xor(rs, m, 32);
      lrow[r] = lrow[r] * alpha + rs;
      mrow[r] = mnew;
#pragma unroll
      for (int t = 0; t < 4; t++) oacc[t][r] *= alpha;
      pl[(rbase + r) * 32 + col]      = (_Float16)p0;
      pl[(rbase + r) * 32 + 16 + col] = (_Float16)p1;
    }
    // reload P in A-fragment layout and accumulate P(16x32) * V(32x64)
    {
      int prow = lane & 15;
      int pk   = (lane >> 4) << 3;
      const _Float16* pp = pl + prow * 32 + pk;
      v8h lo = *(const v8h*)pp;
      v8h hi = *(const v8h*)(pp + 16);
      v16h pa = __builtin_shufflevector(lo, hi, 0,1,2,3,4,5,6,7,8,9,10,11,12,13,14,15);
#pragma unroll
      for (int t = 0; t < 4; t++) {
        v16h vf = load_frag(vb, S, t * 16, jb);
        oacc[t] = WMMA_F16(pa, vf, oacc[t]);
      }
    }
  }

  const int b = bh >> 4, h = bh & 15;
#pragma unroll
  for (int r = 0; r < 8; r++) {
    float inv = 1.f / lrow[r];
    int srow = qrow0 + rbase + r;
    _Float16* orow = o + ((size_t)(b * 1024 + srow)) * 1024 + h * 64;
#pragma unroll
    for (int t = 0; t < 4; t++) orow[t * 16 + col] = (_Float16)(oacc[t][r] * inv);
  }
}

// ---------------------------------------------------------------------------
// Host-side orchestration
// ---------------------------------------------------------------------------
extern "C" void kernel_launch(void* const* d_in, const int* in_sizes, int n_in,
                              void* d_out, int out_size, void* d_ws, size_t ws_size,
                              hipStream_t stream) {
  (void)in_sizes; (void)n_in; (void)out_size; (void)ws_size;
  const float* x      = (const float*)d_in[0];   // (4,1024,1024)
  const float* w_in   = (const float*)d_in[1];   // (3072,1024)
  const float* w_out  = (const float*)d_in[2];   // (1024,1024)
  const float* ln1_g  = (const float*)d_in[3];
  const float* ln1_b  = (const float*)d_in[4];
  const float* w_up   = (const float*)d_in[5];   // (1024,2048)
  const float* b_up   = (const float*)d_in[6];
  const float* w_down = (const float*)d_in[7];   // (2048,1024)
  const float* b_down = (const float*)d_in[8];
  const float* ln2_g  = (const float*)d_in[9];
  const float* ln2_b  = (const float*)d_in[10];
  float* out = (float*)d_out;

  const int BS = 4096, D = 1024;

  char* ws = (char*)d_ws;
  size_t off = 0;
  auto alloc = [&](size_t bytes) -> void* {
    off = (off + 255) & ~(size_t)255;
    void* p = ws + off;
    off += bytes;
    return p;
  };

  _Float16* w_in16   = (_Float16*)alloc((size_t)3 * D * D * 2);
  _Float16* w_outT   = (_Float16*)alloc((size_t)D * D * 2);
  _Float16* w_upT    = (_Float16*)alloc((size_t)2 * D * D * 2);
  _Float16* w_downT  = (_Float16*)alloc((size_t)2 * D * D * 2);
  _Float16* h16      = (_Float16*)alloc((size_t)BS * D * 2);
  _Float16* qkv16    = (_Float16*)alloc((size_t)BS * 3 * D * 2);
  _Float16* q16      = (_Float16*)alloc((size_t)BS * D * 2);
  _Float16* k16      = (_Float16*)alloc((size_t)BS * D * 2);
  _Float16* vT16     = (_Float16*)alloc((size_t)BS * D * 2);
  _Float16* attn16   = (_Float16*)alloc((size_t)BS * D * 2);
  float*    x1       = (float*)   alloc((size_t)BS * D * 4);
  _Float16* h2_16    = (_Float16*)alloc((size_t)BS * D * 2);
  _Float16* ff16     = (_Float16*)alloc((size_t)BS * 2 * D * 2);

  // --- weight conversion ---
  cvt_f32_to_f16<<<(3 * D * D + 255) / 256, 256, 0, stream>>>(w_in, w_in16, 3 * D * D);
  transpose_to_f16<<<(D * D + 255) / 256, 256, 0, stream>>>(w_out, w_outT, D, D);
  transpose_to_f16<<<(2 * D * D + 255) / 256, 256, 0, stream>>>(w_up, w_upT, D, 2 * D);
  transpose_to_f16<<<(2 * D * D + 255) / 256, 256, 0, stream>>>(w_down, w_downT, 2 * D, D);

  // --- LN1 -> h16 ---
  layernorm_to_f16<<<BS, 256, 0, stream>>>(x, ln1_g, ln1_b, h16);

  // --- qkv = h @ in_project.T  (M=4096, N=3072, K=1024), f16 out ---
  gemm_wmma<false, false, false, false, true>
      <<<dim3(3072 / 128, BS / 128), 256, 0, stream>>>(
          h16, w_in16, nullptr, nullptr, nullptr, qkv16, BS, 3 * D, D);

  // --- scatter to q/k (B*H,S,dh) and vT (B*H,dh,S) ---
  reorder_qkv<<<(BS * 3 * D + 255) / 256, 256, 0, stream>>>(qkv16, q16, k16, vT16);

  // --- flash attention -> attn16 (B,S,H*dh) ---
  attention_kernel<<<dim3(1024 / 64, 64), 128, 0, stream>>>(q16, k16, vT16, attn16);

  // --- x1 = attn @ out_project + x  (M=4096, N=1024, K=1024), f32 out ---
  gemm_wmma<false, false, true, true, false>
      <<<dim3(1024 / 128, BS / 128), 256, 0, stream>>>(
          attn16, w_outT, nullptr, x, x1, nullptr, BS, D, D);

  // --- LN2 -> h2_16 ---
  layernorm_to_f16<<<BS, 256, 0, stream>>>(x1, ln2_g, ln2_b, h2_16);

  // --- ff = relu(h2 @ w_up + b_up)  (M=4096, N=2048, K=1024), f16 out ---
  gemm_wmma<true, true, false, false, true>
      <<<dim3(2048 / 128, BS / 128), 256, 0, stream>>>(
          h2_16, w_upT, b_up, nullptr, nullptr, ff16, BS, 2 * D, D);

  // --- out = ff @ w_down + b_down + x1  (M=4096, N=1024, K=2048), f32 out ---
  gemm_wmma<true, false, true, true, false>
      <<<dim3(1024 / 128, BS / 128), 256, 0, stream>>>(
          ff16, w_downT, b_down, x1, out, nullptr, BS, D, 2 * D);
}
